// GlobalRouters_76742475645439
// MI455X (gfx1250) — compile-verified, matrix operand order checked by reading, not run on hardware
//
#include <hip/hip_runtime.h>
#include <hip/hip_bf16.h>

typedef __attribute__((ext_vector_type(16))) _Float16 v16h;
typedef __attribute__((ext_vector_type(8)))  float    v8f;
typedef __attribute__((ext_vector_type(4)))  float    v4f;

#define D_MODEL   2048
#define NTOT      160      // 64 (compress) + 32 + 32 + 32
#define B_BATCH   4
#define S_SEQ     4096
#define ROWS      (B_BATCH * S_SEQ)   // 16384
#define KSTEPS    (D_MODEL / 32)      // 64
#define NCTILES   (NTOT / 16)         // 10
#define WAVES_PB  4
#define ROWS_PB   (WAVES_PB * 16)     // 64 rows per block
#define LDS_STRIDE 161                // pad 160 -> 161 to break bank conflicts

// ---------------------------------------------------------------------------
// Kernel 0: zero the dense accumulator [B][NTOT]
// ---------------------------------------------------------------------------
__global__ void zero_dense_kernel(float* __restrict__ dense) {
    int i = threadIdx.x;
    if (i < B_BATCH * NTOT) dense[i] = 0.0f;
}

// ---------------------------------------------------------------------------
// Kernel 1: convert W (fp32, [K, N] row-major per router) into f16 WMMA
// B-fragment layout: wfrag[((ks*NCTILES + c)*32 + lane)*16 + j]
// lane 0-15  : col = c*16 + lane,     K halves j<8 -> K=j,   j>=8 -> K=8+j
// lane 16-31 : col = c*16 + (lane-16), K halves j<8 -> K=8+j, j>=8 -> K=16+j
// (matches the ISA 16-bit 16x32 fragment table)
// ---------------------------------------------------------------------------
__global__ void convert_w_kernel(const float* __restrict__ Wc,
                                 const float* __restrict__ Wq,
                                 const float* __restrict__ Wk,
                                 const float* __restrict__ Wv,
                                 _Float16* __restrict__ wfrag) {
    const int blk  = blockIdx.x;           // 0 .. KSTEPS*NCTILES-1
    const int ks   = blk / NCTILES;
    const int c    = blk % NCTILES;
    const int lane = threadIdx.x & 31;

    const int n = c * 16 + (lane & 15);    // global fused column 0..159
    const float* W; int ld; int ncol;
    if (n < 64)       { W = Wc; ld = 64; ncol = n;       }
    else if (n < 96)  { W = Wq; ld = 32; ncol = n - 64;  }
    else if (n < 128) { W = Wk; ld = 32; ncol = n - 96;  }
    else              { W = Wv; ld = 32; ncol = n - 128; }

    _Float16* out = wfrag + (((size_t)ks * NCTILES + c) * 32 + lane) * 16;
    const int kb = ks * 32 + ((lane >> 4) * 8);
#pragma unroll
    for (int j = 0; j < 16; ++j) {
        const int k = kb + (j < 8 ? j : j + 8);
        out[j] = (_Float16)W[(size_t)k * ld + ncol];
    }
}

// ---------------------------------------------------------------------------
// softmax over lrow[off .. off+n) scaled by importance, accumulated into
// the block-level LDS partial via ds_add_f32
// ---------------------------------------------------------------------------
__device__ __forceinline__ void softmax_accum(const float* lrow, int off, int n,
                                              float impv, float* partial) {
    float m = -3.4e38f;
    for (int j = 0; j < n; ++j) m = fmaxf(m, lrow[off + j]);
    float s = 0.0f;
    for (int j = 0; j < n; ++j) s += __expf(lrow[off + j] - m);
    const float scale = impv / s;
    for (int j = 0; j < n; ++j)
        atomicAdd(&partial[off + j], __expf(lrow[off + j] - m) * scale);
}

// ---------------------------------------------------------------------------
// Kernel 2: fused router GEMM (f16 WMMA) + per-row segment softmax +
// importance-weighted accumulation into dense[B][NTOT]
// 128 threads = 4 waves; wave w owns rows [blk*64 + w*16, +16), all 160 cols
// ---------------------------------------------------------------------------
__global__ __launch_bounds__(WAVES_PB * 32)
void router_gemm_kernel(const float* __restrict__ x,
                        const float* __restrict__ imp,
                        const _Float16* __restrict__ wfrag,
                        float* __restrict__ dense) {
    __shared__ float lds_logits[WAVES_PB][16][LDS_STRIDE];
    __shared__ float partial[NTOT];

    const int tid  = threadIdx.x;
    const int wave = tid >> 5;
    const int lane = tid & 31;

    for (int i = tid; i < NTOT; i += blockDim.x) partial[i] = 0.0f;
    __syncthreads();

    const int rowBase = blockIdx.x * ROWS_PB + wave * 16;
    const int myRow   = rowBase + (lane & 15);
    const int kSel    = (lane >> 4) * 8;      // A-fragment K sub-offset per ISA layout

    v8f acc[NCTILES] = {};

    const float* xRow = x + (size_t)myRow * D_MODEL;

    for (int ks = 0; ks < KSTEPS; ++ks) {
        const int koff = ks * 32 + kSel;
        // A fragment: 16 halves = K {koff..koff+7} and {koff+16..koff+23}
        const v4f x0 = *(const v4f*)(xRow + koff);
        const v4f x1 = *(const v4f*)(xRow + koff + 4);
        const v4f x2 = *(const v4f*)(xRow + koff + 16);
        const v4f x3 = *(const v4f*)(xRow + koff + 20);
        v16h a;
#pragma unroll
        for (int j = 0; j < 4; ++j) {
            a[j]      = (_Float16)x0[j];
            a[4 + j]  = (_Float16)x1[j];
            a[8 + j]  = (_Float16)x2[j];
            a[12 + j] = (_Float16)x3[j];
        }
        const _Float16* wp = wfrag + ((size_t)ks * NCTILES * 32 + lane) * 16;
#pragma unroll
        for (int c = 0; c < NCTILES; ++c) {
            const v16h b = *(const v16h*)(wp + (size_t)c * 32 * 16);
            acc[c] = __builtin_amdgcn_wmma_f32_16x16x32_f16(
                false, a, false, b, (short)0, acc[c], false, false);
        }
    }

    // Spill logits: C/D layout -> row = (lane>=16 ? 8 : 0) + vgpr_i, col = c*16 + (lane&15)
    const int rOff = (lane >> 4) * 8;
#pragma unroll
    for (int c = 0; c < NCTILES; ++c) {
        const int nloc = c * 16 + (lane & 15);
#pragma unroll
        for (int i = 0; i < 8; ++i)
            lds_logits[wave][rOff + i][nloc] = acc[c][i];
    }
    __syncthreads();

    // Per-row segment softmax: lanes 0-15 handle the 64-wide compress segment,
    // lanes 16-31 handle the three 32-wide expand segments of the same row.
    {
        const int r    = lane & 15;
        const int grow = rowBase + r;             // == b*4096 + s
        const float impv = imp[grow];
        const float* lrow = &lds_logits[wave][r][0];
        if (lane < 16) {
            softmax_accum(lrow, 0, 64, impv, partial);
        } else {
            softmax_accum(lrow, 64, 32, impv, partial);
            softmax_accum(lrow, 96, 32, impv, partial);
            softmax_accum(lrow, 128, 32, impv, partial);
        }
    }
    __syncthreads();

    // One global atomic per (block, column); all 64 rows of a block share b.
    const int b = (blockIdx.x * ROWS_PB) / S_SEQ;
    for (int i = tid; i < NTOT; i += blockDim.x)
        atomicAdd(&dense[b * NTOT + i], partial[i]);
}

// ---------------------------------------------------------------------------
// Kernel 3: top-k sparsify + renormalize. 16 tasks = (b, router).
// Output tuple order: cw[4][64], qw[4][32], kw[4][32], vw[4][32]
// ---------------------------------------------------------------------------
__global__ void finalize_kernel(const float* __restrict__ dense,
                                float* __restrict__ out) {
    if (threadIdx.x != 0) return;
    const int t = blockIdx.x;        // 0..15
    const int b = t >> 2, r = t & 3;
    int colOff, N, K, outBase;
    switch (r) {
        case 0:  colOff = 0;   N = 64; K = 8; outBase = b * 64;        break;
        case 1:  colOff = 64;  N = 32; K = 4; outBase = 256 + b * 32;  break;
        case 2:  colOff = 96;  N = 32; K = 4; outBase = 384 + b * 32;  break;
        default: colOff = 128; N = 32; K = 4; outBase = 512 + b * 32;  break;
    }
    const float* v = dense + b * NTOT + colOff;

    unsigned long long sel = 0ull;
    float sum = 0.0f;
    for (int it = 0; it < K; ++it) {
        float best = -3.4e38f; int bi = 0;
        for (int n = 0; n < N; ++n) {
            if ((sel >> n) & 1ull) continue;
            if (v[n] > best) { best = v[n]; bi = n; }
        }
        sel |= 1ull << bi;
        sum += best;
    }
    const float inv = 1.0f / (sum + 1e-8f);
    for (int n = 0; n < N; ++n)
        out[outBase + n] = ((sel >> n) & 1ull) ? v[n] * inv : 0.0f;
}

// ---------------------------------------------------------------------------
extern "C" void kernel_launch(void* const* d_in, const int* in_sizes, int n_in,
                              void* d_out, int out_size, void* d_ws, size_t ws_size,
                              hipStream_t stream) {
    const float* x   = (const float*)d_in[0];   // [4,4096,2048]
    const float* imp = (const float*)d_in[1];   // [4,4096]
    const float* Wc  = (const float*)d_in[2];   // [2048,64]
    const float* Wq  = (const float*)d_in[3];   // [2048,32]
    const float* Wk  = (const float*)d_in[4];   // [2048,32]
    const float* Wv  = (const float*)d_in[5];   // [2048,32]
    float* out = (float*)d_out;                 // 640 floats

    const size_t wfragBytes = (size_t)KSTEPS * NCTILES * 32 * 16 * sizeof(_Float16); // 640 KB
    _Float16* wfrag = (_Float16*)d_ws;
    float*    dense = (float*)((char*)d_ws + wfragBytes);   // [4][160]

    zero_dense_kernel<<<1, 1024, 0, stream>>>(dense);
    convert_w_kernel<<<KSTEPS * NCTILES, 32, 0, stream>>>(Wc, Wq, Wk, Wv, wfrag);
    router_gemm_kernel<<<ROWS / ROWS_PB, WAVES_PB * 32, 0, stream>>>(x, imp, wfrag, dense);
    finalize_kernel<<<16, 32, 0, stream>>>(dense, out);
}
